// AttentionRNN_472446402786
// MI455X (gfx1250) — matrix operand order repaired and checked by reading, not compiled, not run on hardware
//
#include <hip/hip_runtime.h>
#include <hip/hip_bf16.h>

typedef __attribute__((ext_vector_type(16))) __bf16 v16bf;
typedef __attribute__((ext_vector_type(8)))  float  v8f;

#define INPUT_SZ 256
#define HIDDEN   512
#define OUTPUT   256
#define BATCH    64
#define SEQ      1024

__device__ __forceinline__ float fast_tanh(float x) {
#if __has_builtin(__builtin_amdgcn_tanhf)
  return __builtin_amdgcn_tanhf(x);
#else
  float e = __expf(-2.0f * x);
  return (1.0f - e) / (1.0f + e);
#endif
}

// ---------------------------------------------------------------------------
// Pack W rows (row-major, leading dim ldw, starting column col0) into the
// WMMA bf16 B-tile layout.  B tile (kt, tn) is 32(K) x 16(N):
//   lane l holds column n = tn*16 + (l&15);
//   halves hv=0..7 hold K = kt*32 + (l>>4)*8 + hv; hv=8..15 hold that + 8.
// Tile storage: out[((tn*KT + kt)*32 + lane)*16 + hv]
// ---------------------------------------------------------------------------
__global__ void pack_b_kernel(const float* __restrict__ W, int ldw, int col0,
                              __bf16* __restrict__ out, int KT) {
  int idx  = blockIdx.x * blockDim.x + threadIdx.x;
  int lane = idx & 31;
  int tile = idx >> 5;
  int kt   = tile % KT;
  int n    = (tile / KT) * 16 + (lane & 15);
  int k0   = kt * 32 + ((lane >> 4) << 3);
  const float* src = W + (size_t)n * ldw + col0 + k0;
  __bf16* dst = out + ((size_t)tile * 32 + lane) * 16;
#pragma unroll
  for (int i = 0; i < 16; ++i) dst[i] = (__bf16)src[i < 8 ? i : i + 8];
}

// ---------------------------------------------------------------------------
// Pack x (f32, [b][s][i]) into WMMA bf16 A-tile layout, rows r = s*64+b.
// A tile (tm, kt) is 16(M) x 32(K), ISA 16-bit A layout:
//   lanes 0-15: M = lane, K = {0..7, 16..23}; lanes 16-31: same M, K + 8.
// Tile storage: xab[((tm*8 + kt)*32 + lane)*16 + hv]
// ---------------------------------------------------------------------------
__global__ void pack_a_kernel(const float* __restrict__ x,
                              __bf16* __restrict__ xab) {
  int idx  = blockIdx.x * blockDim.x + threadIdx.x;   // 4096*8*32 threads
  int lane = idx & 31;
  int tile = idx >> 5;                                // tm*8 + kt
  int kt   = tile & 7;
  int tm   = tile >> 3;
  int m    = lane & 15;
  int kb   = (lane >> 4) << 3;
  int row  = tm * 16 + m;                             // r = s*64 + b
  int s = row >> 6, b = row & 63;
  const float* src = x + ((size_t)b * SEQ + s) * INPUT_SZ + kt * 32 + kb;
  __bf16* dst = xab + ((size_t)tile * 32 + lane) * 16;
  const int koff[8] = {0, 2, 4, 6, 16, 18, 20, 22};
#pragma unroll
  for (int v = 0; v < 8; ++v) {
    float2 f2 = *(const float2*)(src + koff[v]);
    dst[2 * v]     = (__bf16)f2.x;
    dst[2 * v + 1] = (__bf16)f2.y;
  }
}

// ---------------------------------------------------------------------------
// Phase 1: x_proj[r][h] = x @ Wx^T + b_ih  (M=65536, N=512, K=256), bf16 out.
// One wave computes a 16x64 strip: 1 A tile reused across 4 N tiles.
// ---------------------------------------------------------------------------
__global__ void xproj_kernel(const __bf16* __restrict__ xab,
                             const __bf16* __restrict__ Wxb,
                             const float* __restrict__ bih,
                             __bf16* __restrict__ xpb) {
  const int lane = threadIdx.x & 31;
  const int wave = threadIdx.x >> 5;
  const int gid  = blockIdx.x * 8 + wave;   // 32768 wave-jobs
  const int tm   = gid >> 3;                // M tile (0..4095)
  const int tn0  = (gid & 7) * 4;           // first of 4 N tiles

  v8f acc0 = {}, acc1 = {}, acc2 = {}, acc3 = {};
#pragma unroll
  for (int kt = 0; kt < 8; ++kt) {
    v16bf a = *(const v16bf*)(xab + (((size_t)tm * 8 + kt) * 32 + lane) * 16);
    v16bf b0 = *(const v16bf*)(Wxb + (((size_t)(tn0 + 0) * 8 + kt) * 32 + lane) * 16);
    v16bf b1 = *(const v16bf*)(Wxb + (((size_t)(tn0 + 1) * 8 + kt) * 32 + lane) * 16);
    v16bf b2 = *(const v16bf*)(Wxb + (((size_t)(tn0 + 2) * 8 + kt) * 32 + lane) * 16);
    v16bf b3 = *(const v16bf*)(Wxb + (((size_t)(tn0 + 3) * 8 + kt) * 32 + lane) * 16);
    acc0 = __builtin_amdgcn_wmma_f32_16x16x32_bf16(false, a, false, b0, (short)0, acc0, false, false);
    acc1 = __builtin_amdgcn_wmma_f32_16x16x32_bf16(false, a, false, b1, (short)0, acc1, false, false);
    acc2 = __builtin_amdgcn_wmma_f32_16x16x32_bf16(false, a, false, b2, (short)0, acc2, false, false);
    acc3 = __builtin_amdgcn_wmma_f32_16x16x32_bf16(false, a, false, b3, (short)0, acc3, false, false);
  }
  const int mlo = lane & 15;
  const int mhi = (lane >> 4) << 3;
#pragma unroll
  for (int j = 0; j < 4; ++j) {
    v8f acc = (j == 0) ? acc0 : (j == 1) ? acc1 : (j == 2) ? acc2 : acc3;
    int col = (tn0 + j) * 16 + mlo;
    float bv = bih[col];
#pragma unroll
    for (int v = 0; v < 8; ++v) {
      int r = tm * 16 + v + mhi;
      xpb[(size_t)r * HIDDEN + col] = (__bf16)(acc[v] + bv);
    }
  }
}

// ---------------------------------------------------------------------------
// Phase 2: recurrence.  4 blocks (batch groups of 16 rows), 1024 threads
// (32 waves).  Wave w keeps its 16 columns of Wh^T (16 K-tiles) in VGPRs for
// all 1024 steps.  h double-buffered in LDS in WMMA A-layout.  The per-step
// x_proj tile (16x512 bf16 = 16 KB) is DMA'd one step ahead into an LDS
// double buffer via global_load_async_to_lds_b128 (ASYNCcnt).
// ---------------------------------------------------------------------------
__global__ void rnn_kernel(const __bf16* __restrict__ xpb,
                           const __bf16* __restrict__ Whb,
                           float* __restrict__ hs) {
  __shared__ __bf16 hbuf[2 * 16 * 32 * 16];   // 32 KB: 2 x [kt][lane][hv]
  __shared__ __bf16 xpbuf[2 * 16 * HIDDEN];   // 32 KB: 2 x [m][col]
  const int tid  = threadIdx.x;
  const int lane = tid & 31;
  const int tn   = tid >> 5;                  // wave id == N tile
  const int bg   = blockIdx.x;                // batch group (0..3)

  // Resident B: this wave's 16 columns of Wh^T across full K (128 VGPRs).
  v16bf Breg[16];
#pragma unroll
  for (int kt = 0; kt < 16; ++kt)
    Breg[kt] = *(const v16bf*)(Whb + (((size_t)tn * 16 + kt) * 32 + lane) * 16);

  // Async copy helper state: each thread moves 16 bytes (8 bf16) per step.
  const uint32_t xp_lds0 = (uint32_t)(uintptr_t)(&xpbuf[0]) + (uint32_t)tid * 16u;
  const __bf16* xp_g0 = xpb + (size_t)bg * 16 * HIDDEN + (size_t)tid * 8;

  // Prologue: kick off tile t=0 into buffer 0.
  {
    const __bf16* g = xp_g0;                  // t = 0
    asm volatile("global_load_async_to_lds_b128 %0, %1, off"
                 :: "v"(xp_lds0), "v"(g) : "memory");
  }

  // h0 = 0 (zero buffer 0: 16 KB = 4096 dwords, 4 per thread)
  {
    uint32_t* z = (uint32_t*)hbuf;
#pragma unroll
    for (int i = 0; i < 4; ++i) z[tid * 4 + i] = 0u;
  }
  __syncthreads();

  // Per-lane constants for writing h_new back into A-layout.
  const int laneHi = lane >> 4;
  const int n      = lane & 15;
  const int col    = tn * 16 + n;                 // global hidden index
  const int klocal = ((tn & 1) << 4) | n;         // K position inside A k-tile
  const int g      = (klocal >> 3) & 1;           // lane-half group
  const int r0     = klocal - (g << 3);           // {0..7} u {16..23}
  const int hv     = (r0 < 8) ? r0 : (r0 - 8);    // half index within lane
  const int kt2    = tn >> 1;                     // destination A k-tile

  for (int t = 0; t < SEQ; ++t) {
    const int rd = t & 1;
    const int wr = rd ^ 1;

    // Kick off DMA of x_proj tile t+1 into the other xp buffer.
    if (t + 1 < SEQ) {
      const __bf16* gp = xp_g0 + (size_t)(t + 1) * BATCH * HIDDEN;
      uint32_t dst = xp_lds0 + (uint32_t)wr * (16 * HIDDEN * 2);
      asm volatile("global_load_async_to_lds_b128 %0, %1, off"
                   :: "v"(dst), "v"(gp) : "memory");
    }

    // GEMM: 16x512x512 strip, B resident in VGPRs, A broadcast from LDS.
    const __bf16* hrd = hbuf + rd * (16 * 32 * 16);
    v8f acc = {};
#pragma unroll
    for (int kt = 0; kt < 16; ++kt) {
      v16bf a = *(const v16bf*)(hrd + ((size_t)kt * 32 + lane) * 16);
      acc = __builtin_amdgcn_wmma_f32_16x16x32_bf16(false, a, false, Breg[kt],
                                                    (short)0, acc, false, false);
    }

    // Drain this wave's async copy of tile t (in-order: <=1 leaves only t+1).
    if (t + 1 < SEQ) {
      asm volatile("s_wait_asynccnt 0x1" ::: "memory");
    } else {
      asm volatile("s_wait_asynccnt 0x0" ::: "memory");
    }
    __syncthreads();   // publish tile t across waves

    const __bf16* xpt = xpbuf + rd * (16 * HIDDEN);
    __bf16* hwr = hbuf + wr * (16 * 32 * 16) + (size_t)kt2 * 32 * 16;
#pragma unroll
    for (int v = 0; v < 8; ++v) {
      int   m    = v + (laneHi << 3);
      float hval = fast_tanh(acc[v] + (float)xpt[m * HIDDEN + col]);
      int   b    = bg * 16 + m;
      hs[((size_t)b * SEQ + t) * HIDDEN + col] = hval;     // f32 for attention
      int dl = m + (g << 4);
      hwr[dl * 16 + hv] = (__bf16)hval;                    // bf16 A-layout
    }
    __syncthreads();   // publish h(t+1) before next step's GEMM
  }
}

// ---------------------------------------------------------------------------
// Phase 3: attention + output projection.  One block per batch row.
// ---------------------------------------------------------------------------
__global__ void attn_out_kernel(const float* __restrict__ hs,
                                const float* __restrict__ Who,
                                const float* __restrict__ bho,
                                float* __restrict__ out) {
  __shared__ float fh[HIDDEN];
  __shared__ float sc[SEQ];
  __shared__ float red[256];
  __shared__ float ctx[HIDDEN];
  const int b   = blockIdx.x;
  const int tid = threadIdx.x;
  const float* hsb = hs + (size_t)b * SEQ * HIDDEN;

#pragma unroll
  for (int i = 0; i < 2; ++i)
    fh[tid + 256 * i] = hsb[(size_t)(SEQ - 1) * HIDDEN + tid + 256 * i];
  __syncthreads();

  float lmax = -3.4e38f;
#pragma unroll
  for (int j = 0; j < 4; ++j) {
    int s = tid + 256 * j;
    const float* row = hsb + (size_t)s * HIDDEN;
    float acc = 0.f;
    for (int h = 0; h < HIDDEN; ++h) acc += row[h] * fh[h];
    sc[s] = acc;
    lmax  = fmaxf(lmax, acc);
  }
  red[tid] = lmax;
  __syncthreads();
  for (int st = 128; st > 0; st >>= 1) {
    if (tid < st) red[tid] = fmaxf(red[tid], red[tid + st]);
    __syncthreads();
  }
  const float gmax = red[0];
  __syncthreads();

  float lsum = 0.f;
#pragma unroll
  for (int j = 0; j < 4; ++j) {
    int s   = tid + 256 * j;
    float e = __expf(sc[s] - gmax);
    sc[s]   = e;
    lsum   += e;
  }
  red[tid] = lsum;
  __syncthreads();
  for (int st = 128; st > 0; st >>= 1) {
    if (tid < st) red[tid] += red[tid + st];
    __syncthreads();
  }
  const float inv = 1.0f / red[0];
  __syncthreads();

#pragma unroll
  for (int i = 0; i < 2; ++i) {
    int h = tid + 256 * i;
    float c = 0.f;
    for (int s = 0; s < SEQ; ++s) c += sc[s] * hsb[(size_t)s * HIDDEN + h];
    ctx[h] = c * inv;
  }
  __syncthreads();

  {
    int o = tid;
    const float* wrow = Who + (size_t)o * HIDDEN;
    float acc = bho[o];
    for (int h = 0; h < HIDDEN; ++h) acc += ctx[h] * wrow[h];
    out[(size_t)b * OUTPUT + o] = acc;
  }
}

// ---------------------------------------------------------------------------
extern "C" void kernel_launch(void* const* d_in, const int* in_sizes, int n_in,
                              void* d_out, int out_size, void* d_ws, size_t ws_size,
                              hipStream_t stream) {
  (void)in_sizes; (void)n_in; (void)out_size; (void)ws_size;
  const float* x    = (const float*)d_in[0];
  const float* W_ih = (const float*)d_in[1];
  const float* b_ih = (const float*)d_in[2];
  const float* W_ho = (const float*)d_in[3];
  const float* b_ho = (const float*)d_in[4];
  float* out = (float*)d_out;

  // Workspace carve-up (~225 MB):
  float*  hs  = (float*)d_ws;                                   // [64][1024][512] f32
  __bf16* xpb = (__bf16*)(hs + (size_t)BATCH * SEQ * HIDDEN);   // [65536][512] bf16
  __bf16* xab = xpb + (size_t)BATCH * SEQ * HIDDEN;             // packed x A-tiles bf16
  __bf16* wxb = xab + (size_t)BATCH * SEQ * INPUT_SZ;           // packed Wx bf16
  __bf16* whb = wxb + (size_t)HIDDEN * INPUT_SZ;                // packed Wh bf16

  // Pack weights into B-tile layout.
  pack_b_kernel<<<32, 256, 0, stream>>>(W_ih, INPUT_SZ + HIDDEN, 0, wxb, INPUT_SZ / 32);
  pack_b_kernel<<<64, 256, 0, stream>>>(W_ih, INPUT_SZ + HIDDEN, INPUT_SZ, whb, HIDDEN / 32);

  // Pack x into A-tile layout (one-time f32->bf16).
  pack_a_kernel<<<4096, 256, 0, stream>>>(x, xab);

  // Phase 1: 4096 M-tiles x 8 N-groups, 8 waves/block.
  xproj_kernel<<<4096, 256, 0, stream>>>(xab, wxb, b_ih, xpb);

  // Phase 2: 4 batch groups, 32 waves each, no cross-WG sync needed.
  rnn_kernel<<<4, 1024, 0, stream>>>(xpb, whb, hs);

  // Phase 3: attention + output.
  attn_out_kernel<<<64, 256, 0, stream>>>(hs, W_ho, b_ho, out);
}
